// GDCGraphCL_9646496547162
// MI455X (gfx1250) — matrix operand order repaired and checked by reading, not compiled, not run on hardware
//
#include <hip/hip_runtime.h>
#include <hip/hip_bf16.h>

// ---------------------------------------------------------------------------
// Problem constants (match reference)
// ---------------------------------------------------------------------------
#define NN     50000
#define EE     800000
#define IN_D   256
#define HID    512
#define OUT_D  64
#define PROJ   64
#define MAX_K  512   // largest GEMM K dimension (HID)

typedef __attribute__((ext_vector_type(16))) __bf16 v16bf;
typedef __attribute__((ext_vector_type(8)))  float  v8f;

// ---------------------------------------------------------------------------
// WMMA GEMM:  C[M,Nc] = act( A[M,K] @ W[K,Nc] + bias )
//   act: 0 = none, 1 = relu, 2 = prelu(alpha[0])
//
// Block = 128 threads (4 waves) covering 4 M-tiles x 1 N-tile (64x16 output).
// The 16-column weight panel W[:,col0:col0+16] is staged ONCE per block into
// LDS as packed bf16 pairs (k even/odd packed in one dword), so the WMMA loop
// reads B with one ds_load_b32 per fragment VGPR and A with float2 global
// loads converted on the fly. Accumulation in f32 (v_wmma_f32_16x16x32_bf16).
//
// Fragment layouts used here (contiguous K, wave32):
//   A 16x32 : lane<16: M=lane,   K=2i,2i+1           ; lane>=16: M=lane-16, K=16+2i
//   B 32x16 : lane<16: N=lane,   K=2i,2i+1           ; lane>=16: N=lane-16, K=16+2i
//   C/D     : lane L, VGPR r -> (M = r + 8*(L>>4), N = L&15)
// ---------------------------------------------------------------------------
__global__ void wmma_gemm_bias_act(const float* __restrict__ A,
                                   const float* __restrict__ W,
                                   const float* __restrict__ bias,
                                   const float* __restrict__ alpha,
                                   float* __restrict__ C,
                                   int M, int K, int Nc, int act)
{
    __shared__ unsigned Wlds[(MAX_K / 2) * 16];   // packed bf16 pairs, 16 KB max

    const int lane   = threadIdx.x & 31;
    const int waveid = threadIdx.x >> 5;
    const int tile_m = blockIdx.x * 4 + waveid;   // 4 waves / block
    const int tile_n = blockIdx.y;
    const int col0   = tile_n * 16;

    // ---- cooperative stage: W[:, col0:col0+16] -> LDS as packed bf16 ----
    const int totalPk = (K >> 1) * 16;            // one dword per (k-pair, col)
    for (int idx = threadIdx.x; idx < totalPk; idx += blockDim.x) {
        const int p = idx >> 4;                   // k pair index
        const int c = idx & 15;
        const float w0 = W[(size_t)(2 * p)     * Nc + col0 + c];
        const float w1 = W[(size_t)(2 * p + 1) * Nc + col0 + c];
        union { __bf16 h[2]; unsigned u; } pk;
        pk.h[0] = (__bf16)w0;
        pk.h[1] = (__bf16)w1;
        Wlds[idx] = pk.u;
    }
    __syncthreads();

    if (tile_m * 16 >= M) return;                 // uniform per wave (EXEC stays full)

    const int row0 = tile_m * 16;
    const int mrow = lane & 15;
    const int khA  = (lane >> 4) << 3;            // 0 or 8  (unused)
    const int khB  = (lane >> 4) << 4;            // 0 or 16 (B K-pattern)
    const int bcol = col0 + mrow;

    const float* __restrict__ arow = A + (size_t)(row0 + mrow) * K;

    v8f acc = {};
#pragma unroll 2
    for (int k0 = 0; k0 < K; k0 += 32) {
        v16bf af, bf;
        const int kp = (k0 + khB) >> 1;           // packed LDS row base
#pragma unroll
        for (int i = 0; i < 8; ++i) {
            const int kA = k0 + khB + (i << 1);
            const float2 a2 = *(const float2*)(arow + kA);
            af[2 * i]     = (__bf16)a2.x;
            af[2 * i + 1] = (__bf16)a2.y;

            union { unsigned u; __bf16 h[2]; } q;
            q.u = Wlds[(kp + i) * 16 + mrow];
            bf[2 * i]     = q.h[0];
            bf[2 * i + 1] = q.h[1];
        }
        acc = __builtin_amdgcn_wmma_f32_16x16x32_bf16(
            false, af, false, bf, (short)0, acc, false, false);
    }

    const float bv = bias ? bias[bcol] : 0.0f;
    const float al = (act == 2) ? alpha[0] : 0.0f;
    const int rA = (lane >> 4) << 3;
#pragma unroll
    for (int r = 0; r < 8; ++r) {
        float v = acc[r] + bv;
        if (act == 1)      v = fmaxf(v, 0.0f);
        else if (act == 2) v = (v >= 0.0f) ? v : al * v;
        C[(size_t)(row0 + r + rA) * Nc + bcol] = v;
    }
}

// ---------------------------------------------------------------------------
// SpMM scatter: out[src[e], c] += val[e] * x[prm(dst[e]), c]
// One thread per (edge, channel); accumulators are L2-resident -> fast atomics.
// D = 1<<Dshift channels.
// ---------------------------------------------------------------------------
__global__ void spmm_scatter(const int* __restrict__ src,
                             const int* __restrict__ dst,
                             const float* __restrict__ val,
                             const float* __restrict__ x,
                             const int* __restrict__ perm,
                             float* __restrict__ out,
                             int E, int Dshift)
{
    const long long idx = (long long)blockIdx.x * blockDim.x + threadIdx.x;
    const long long total = (long long)E << Dshift;
    if (idx >= total) return;
    const int e = (int)(idx >> Dshift);
    const int c = (int)(idx & ((1 << Dshift) - 1));
    int d = dst[e];
    if (perm) d = perm[d];
    const float contrib = val[e] * x[((size_t)d << Dshift) + c];
    atomicAdd(out + ((size_t)src[e] << Dshift) + c, contrib);
}

__global__ void deg_scatter(const int* __restrict__ src,
                            const float* __restrict__ val,
                            float* __restrict__ deg, int E)
{
    const int e = blockIdx.x * blockDim.x + threadIdx.x;
    if (e < E) atomicAdd(deg + src[e], val[e]);
}

// ---------------------------------------------------------------------------
// Elementwise kernels
// ---------------------------------------------------------------------------
__global__ void relu_inplace(float* __restrict__ x, long long n)
{
    const long long i = (long long)blockIdx.x * blockDim.x + threadIdx.x;
    if (i < n) x[i] = fmaxf(x[i], 0.0f);
}

// y = sigmoid(acc / max(deg,1)), in place; D = 1<<Dshift
__global__ void sig_avg_inplace(float* __restrict__ acc,
                                const float* __restrict__ deg,
                                long long n, int Dshift)
{
    const long long i = (long long)blockIdx.x * blockDim.x + threadIdx.x;
    if (i >= n) return;
    const float dg = fmaxf(deg[i >> Dshift], 1.0f);
    const float y = acc[i] / dg;
    acc[i] = 1.0f / (1.0f + __expf(-y));
}

// out[n*stride] = dot(a[n,:64], b[n,:64]) + bb[0]   (one wave per row)
__global__ void rowdot64(const float* __restrict__ a,
                         const float* __restrict__ b,
                         const float* __restrict__ bb,
                         float* __restrict__ out, int n, int stride)
{
    const int wave = (blockIdx.x * blockDim.x + threadIdx.x) >> 5;
    const int lane = threadIdx.x & 31;
    if (wave >= n) return;
    const float* ar = a + (size_t)wave * 64;
    const float* br = b + (size_t)wave * 64;
    float s = ar[lane] * br[lane] + ar[lane + 32] * br[lane + 32];
#pragma unroll
    for (int off = 16; off; off >>= 1) s += __shfl_xor(s, off, 32);
    if (lane == 0) out[(size_t)wave * stride] = s + bb[0];
}

// ---------------------------------------------------------------------------
// Host-side orchestration
// ---------------------------------------------------------------------------
extern "C" void kernel_launch(void* const* d_in, const int* in_sizes, int n_in,
                              void* d_out, int out_size, void* d_ws, size_t ws_size,
                              hipStream_t stream)
{
    const float* feat      = (const float*)d_in[0];
    const int*   adj_src   = (const int*)d_in[1];
    const int*   adj_dst   = (const int*)d_in[2];
    const float* adj_val   = (const float*)d_in[3];
    const int*   gdc_src   = (const int*)d_in[4];
    const int*   gdc_dst   = (const int*)d_in[5];
    const float* gdc_val   = (const float*)d_in[6];
    const int*   neigh_src = (const int*)d_in[7];
    const int*   neigh_dst = (const int*)d_in[8];
    const float* neigh_val = (const float*)d_in[9];
    const int*   diff_src  = (const int*)d_in[10];
    const int*   diff_dst  = (const int*)d_in[11];
    const float* diff_val  = (const float*)d_in[12];
    const int*   perm      = (const int*)d_in[13];
    const float* W_enc     = (const float*)d_in[14];
    const float* b_enc     = (const float*)d_in[15];
    const float* W_dec     = (const float*)d_in[16];
    const float* b_dec     = (const float*)d_in[17];
    const float* W1        = (const float*)d_in[18];
    const float* b1        = (const float*)d_in[19];
    const float* prelu_a   = (const float*)d_in[20];
    const float* W3        = (const float*)d_in[21];
    const float* b3        = (const float*)d_in[22];
    const float* W_disc    = (const float*)d_in[23];
    const float* b_disc    = (const float*)d_in[24];

    // ---- workspace carve-up (floats) ----
    float* ws = (float*)d_ws;
    size_t o = 0;
    const size_t N64  = (size_t)NN * 64;
    float* XW     = ws + o; o += N64;            // feat @ W_enc + b_enc
    float* h      = ws + o; o += N64;
    float* h_g    = ws + o; o += N64;
    float* shuf_h = ws + o; o += N64;
    float* hdec   = ws + o; o += (size_t)NN * 256;
    float* mid    = ws + o; o += (size_t)NN * 512;   // reused by all 5 MLPs
    float* z      = ws + o; o += N64;
    float* z_g    = ws + o; o += N64;
    float* shuf_z = ws + o; o += N64;
    float* avg    = ws + o; o += N64;            // reused: neigh then diff
    float* g      = ws + o; o += N64;
    float* g_g    = ws + o; o += N64;
    float* t0     = ws + o; o += N64;            // h @ W_disc scratch
    float* deg    = ws + o; o += (size_t)NN;

    float* emb_out  = (float*)d_out;                       // [N,256]
    float* ret_out  = (float*)d_out + (size_t)NN * 256;    // [N,2]
    float* reta_out = ret_out + (size_t)NN * 2;            // [N,2]

    // ---- helpers ----
    auto gemm = [&](const float* A, const float* Wt, const float* bias,
                    const float* alpha, float* Cout, int M, int K, int Nc, int act) {
        dim3 grid((unsigned)(((M + 15) / 16 + 3) / 4), (unsigned)(Nc / 16));
        wmma_gemm_bias_act<<<grid, 128, 0, stream>>>(A, Wt, bias, alpha, Cout, M, K, Nc, act);
    };
    auto spmm = [&](const int* s, const int* d, const float* v, const float* x,
                    const int* prm, float* outp, int Dshift) {
        const long long total = (long long)EE << Dshift;
        const unsigned blocks = (unsigned)((total + 255) / 256);
        spmm_scatter<<<blocks, 256, 0, stream>>>(s, d, v, x, prm, outp, EE, Dshift);
    };
    auto relu = [&](float* x, long long n) {
        relu_inplace<<<(unsigned)((n + 255) / 256), 256, 0, stream>>>(x, n);
    };
    auto zero = [&](void* p, size_t bytes) { hipMemsetAsync(p, 0, bytes, stream); };

    // ---- 1. shared linear: XW = feat @ W_enc + b_enc ----
    gemm(feat, W_enc, b_enc, nullptr, XW, NN, IN_D, OUT_D, 0);

    // ---- 2. h = relu(spmm(adj, XW)) ----
    zero(h, N64 * 4);
    spmm(adj_src, adj_dst, adj_val, XW, nullptr, h, 6);
    relu(h, (long long)N64);

    // ---- 3. h_g = relu(spmm(gdc, XW)) ----
    zero(h_g, N64 * 4);
    spmm(gdc_src, gdc_dst, gdc_val, XW, nullptr, h_g, 6);
    relu(h_g, (long long)N64);

    // ---- 4. shuf_h = relu(spmm(adj, XW[perm])) (gather through perm) ----
    zero(shuf_h, N64 * 4);
    spmm(adj_src, adj_dst, adj_val, XW, perm, shuf_h, 6);
    relu(shuf_h, (long long)N64);

    // ---- 5./6. emb = relu(spmm(adj, h @ W_dec + b_dec)) ----
    gemm(h, W_dec, b_dec, nullptr, hdec, NN, OUT_D, IN_D, 0);
    zero(emb_out, (size_t)NN * 256 * 4);
    spmm(adj_src, adj_dst, adj_val, hdec, nullptr, emb_out, 8);
    relu(emb_out, (long long)NN * 256);

    // ---- 7. z / z_g / shuf_z = mlp(...) ----
    gemm(h, W1, b1, prelu_a, mid, NN, OUT_D, HID, 2);
    gemm(mid, W3, b3, nullptr, z, NN, HID, PROJ, 0);
    gemm(h_g, W1, b1, prelu_a, mid, NN, OUT_D, HID, 2);
    gemm(mid, W3, b3, nullptr, z_g, NN, HID, PROJ, 0);
    gemm(shuf_h, W1, b1, prelu_a, mid, NN, OUT_D, HID, 2);
    gemm(mid, W3, b3, nullptr, shuf_z, NN, HID, PROJ, 0);

    // ---- 8. g = mlp(sigmoid(sparse_avg(neigh, h))) ----
    zero(avg, N64 * 4);
    zero(deg, (size_t)NN * 4);
    spmm(neigh_src, neigh_dst, neigh_val, h, nullptr, avg, 6);
    deg_scatter<<<(EE + 255) / 256, 256, 0, stream>>>(neigh_src, neigh_val, deg, EE);
    sig_avg_inplace<<<(unsigned)((N64 + 255) / 256), 256, 0, stream>>>(avg, deg, (long long)N64, 6);
    gemm(avg, W1, b1, prelu_a, mid, NN, OUT_D, HID, 2);
    gemm(mid, W3, b3, nullptr, g, NN, HID, PROJ, 0);

    // ---- 9. g_g = mlp(sigmoid(sparse_avg(diff, h_g))) ----
    zero(avg, N64 * 4);
    zero(deg, (size_t)NN * 4);
    spmm(diff_src, diff_dst, diff_val, h_g, nullptr, avg, 6);
    deg_scatter<<<(EE + 255) / 256, 256, 0, stream>>>(diff_src, diff_val, deg, EE);
    sig_avg_inplace<<<(unsigned)((N64 + 255) / 256), 256, 0, stream>>>(avg, deg, (long long)N64, 6);
    gemm(avg, W1, b1, prelu_a, mid, NN, OUT_D, HID, 2);
    gemm(mid, W3, b3, nullptr, g_g, NN, HID, PROJ, 0);

    // ---- 10. discriminators: bilinear(h,c) = rowdot(h @ W_disc, c) + b ----
    const unsigned rd_blocks = (NN * 32 + 255) / 256;   // one wave per row
    gemm(z, W_disc, nullptr, nullptr, t0, NN, PROJ, PROJ, 0);
    rowdot64<<<rd_blocks, 256, 0, stream>>>(t0, g_g, b_disc, ret_out + 0, NN, 2);
    gemm(z_g, W_disc, nullptr, nullptr, t0, NN, PROJ, PROJ, 0);
    rowdot64<<<rd_blocks, 256, 0, stream>>>(t0, g, b_disc, reta_out + 0, NN, 2);
    gemm(shuf_z, W_disc, nullptr, nullptr, t0, NN, PROJ, PROJ, 0);   // shared
    rowdot64<<<rd_blocks, 256, 0, stream>>>(t0, g_g, b_disc, ret_out + 1, NN, 2);
    rowdot64<<<rd_blocks, 256, 0, stream>>>(t0, g, b_disc, reta_out + 1, NN, 2);
}
